// AtgcnCell_78417512890599
// MI455X (gfx1250) — compile-verified, matrix operand order checked by reading, not run on hardware
//
#include <hip/hip_runtime.h>
#include <hip/hip_bf16.h>

#define F_DIM 128
#define B_DIM 4
#define NEG_SLOPE 0.2f

typedef __attribute__((ext_vector_type(16))) __bf16 bf16x16;
typedef __attribute__((ext_vector_type(8)))  float  f32x8;

union FragU { uint4 q[2]; bf16x16 v; };

// ---------------------------------------------------------------------------
// float -> bf16 conversion
// ---------------------------------------------------------------------------
__global__ void cvt_bf16_kernel(__bf16* __restrict__ dst,
                                const float* __restrict__ src, int n) {
    int i = blockIdx.x * blockDim.x + threadIdx.x;
    if (i < n) dst[i] = (__bf16)src[i];
}

__global__ void fill_kernel(float* __restrict__ p, float v, int n) {
    int i = blockIdx.x * blockDim.x + threadIdx.x;
    if (i < n) p[i] = v;
}

// ---------------------------------------------------------------------------
// WMMA GEMM:  C[M x 128] = A[M x K](bf16) * W[128 x K](bf16)^T + bias, act
// act: 0 = none, 1 = sigmoid, 2 = tanh
// block = 256 threads = 8 waves. Each wave computes a 16x64 strip (4 N-tiles),
// reusing the A fragment across 4 WMMAs. Block covers 64 rows x 128 cols.
// ---------------------------------------------------------------------------
__global__ __launch_bounds__(256)
void wmma_gemm_bias_act(const __bf16* __restrict__ A,
                        const __bf16* __restrict__ W,
                        const float*  __restrict__ bias,
                        float* __restrict__ C,
                        int M, int K, int act)
{
    const int lane  = threadIdx.x & 31;
    const int wave  = threadIdx.x >> 5;        // 0..7
    const int row16 = lane & 15;
    const int half  = lane >> 4;               // k-half selector
    const int m0    = blockIdx.x * 64 + (wave >> 1) * 16;
    const int n0    = (wave & 1) * 64;         // 4 n-tiles from here

    int rowA = m0 + row16;
    if (rowA > M - 1) rowA = M - 1;            // clamp (keeps EXEC all-ones)
    const __bf16* Arow = A + (size_t)rowA * K;
    const __bf16* W0   = W + (size_t)(n0 + row16) * K;   // + t*16*K per tile

    f32x8 acc0 = {}, acc1 = {}, acc2 = {}, acc3 = {};
    for (int kb = 0; kb < K; kb += 32) {
        FragU fa;
        // 16-bit A-frag layout: lanes 0-15 hold K = kb..kb+7 (vgpr0-3) and
        // kb+16..kb+23 (vgpr4-7); lanes 16-31 hold kb+8.. and kb+24..
        fa.q[0] = *(const uint4*)(Arow + kb +      half * 8);
        fa.q[1] = *(const uint4*)(Arow + kb + 16 + half * 8);
        __builtin_prefetch(Arow + kb + 32, 0, 3);

        FragU fb0, fb1, fb2, fb3;
        const __bf16* Wr;
        Wr = W0;                         // n-tile 0
        fb0.q[0] = *(const uint4*)(Wr + kb +      half * 8);
        fb0.q[1] = *(const uint4*)(Wr + kb + 16 + half * 8);
        Wr = W0 + (size_t)16 * K;        // n-tile 1
        fb1.q[0] = *(const uint4*)(Wr + kb +      half * 8);
        fb1.q[1] = *(const uint4*)(Wr + kb + 16 + half * 8);
        Wr = W0 + (size_t)32 * K;        // n-tile 2
        fb2.q[0] = *(const uint4*)(Wr + kb +      half * 8);
        fb2.q[1] = *(const uint4*)(Wr + kb + 16 + half * 8);
        Wr = W0 + (size_t)48 * K;        // n-tile 3
        fb3.q[0] = *(const uint4*)(Wr + kb +      half * 8);
        fb3.q[1] = *(const uint4*)(Wr + kb + 16 + half * 8);

        acc0 = __builtin_amdgcn_wmma_f32_16x16x32_bf16(
                   false, fa.v, false, fb0.v, (short)0, acc0, false, false);
        acc1 = __builtin_amdgcn_wmma_f32_16x16x32_bf16(
                   false, fa.v, false, fb1.v, (short)0, acc1, false, false);
        acc2 = __builtin_amdgcn_wmma_f32_16x16x32_bf16(
                   false, fa.v, false, fb2.v, (short)0, acc2, false, false);
        acc3 = __builtin_amdgcn_wmma_f32_16x16x32_bf16(
                   false, fa.v, false, fb3.v, (short)0, acc3, false, false);
    }

    f32x8 accs[4] = { acc0, acc1, acc2, acc3 };
#pragma unroll
    for (int t = 0; t < 4; ++t) {
        const int col = n0 + t * 16 + row16;
        const float bv = bias[col];
        float* Ccol = C + (size_t)m0 * F_DIM + col;
#pragma unroll
        for (int j = 0; j < 8; ++j) {
            int r = half * 8 + j;              // row within tile
            if (m0 + r < M) {
                float v = accs[t][j] + bv;
                if (act == 1)      v = 1.f / (1.f + __expf(-v));
                else if (act == 2) v = tanhf(v);
                Ccol[(size_t)r * F_DIM] = v;
            }
        }
    }
}

// ---------------------------------------------------------------------------
// Edge phase
// ---------------------------------------------------------------------------
__device__ inline void atomicMaxF(float* addr, float v) {
    if (v >= 0.f) atomicMax((int*)addr, __float_as_int(v));
    else          atomicMin((unsigned int*)addr, __float_as_uint(v));
}

// wave per (edge, batch): score = att . leaky_relu(xl[src] + xr[dst])
__global__ __launch_bounds__(256)
void edge_score_kernel(const float* __restrict__ XL,
                       const float* __restrict__ XR,
                       const long long* __restrict__ ei,
                       const float* __restrict__ att,
                       float* __restrict__ ebuf,
                       float* __restrict__ mbuf,
                       int E, int EN, int N)
{
    int widx = blockIdx.x * (blockDim.x >> 5) + (threadIdx.x >> 5);
    int lane = threadIdx.x & 31;
    if (widx >= EN * B_DIM) return;
    int e = widx >> 2;
    int b = widx & 3;
    int src, dst;
    if (e < E) { src = (int)ei[e]; dst = (int)ei[E + e]; }
    else       { src = dst = e - E; }

    const float4 xa = *(const float4*)(XL + ((size_t)b * N + src) * F_DIM + lane * 4);
    const float4 xd = *(const float4*)(XR + ((size_t)b * N + dst) * F_DIM + lane * 4);
    const float4 at = *(const float4*)(att + lane * 4);

    float s = 0.f, v;
    v = xa.x + xd.x; v = v > 0.f ? v : NEG_SLOPE * v; s += at.x * v;
    v = xa.y + xd.y; v = v > 0.f ? v : NEG_SLOPE * v; s += at.y * v;
    v = xa.z + xd.z; v = v > 0.f ? v : NEG_SLOPE * v; s += at.z * v;
    v = xa.w + xd.w; v = v > 0.f ? v : NEG_SLOPE * v; s += at.w * v;

    for (int off = 16; off > 0; off >>= 1) s += __shfl_xor(s, off, 32);

    if (lane == 0) {
        ebuf[widx] = s;
        atomicMaxF(mbuf + (size_t)dst * B_DIM + b, s);
    }
}

__global__ void edge_expsum_kernel(const long long* __restrict__ ei,
                                   float* __restrict__ ebuf,
                                   const float* __restrict__ mbuf,
                                   float* __restrict__ denom,
                                   int E, int EN, int N)
{
    int i = blockIdx.x * blockDim.x + threadIdx.x;
    if (i >= EN * B_DIM) return;
    int e = i >> 2, b = i & 3;
    int dst = (e < E) ? (int)ei[E + e] : (e - E);
    float ex = __expf(ebuf[i] - mbuf[(size_t)dst * B_DIM + b]);
    ebuf[i] = ex;
    atomicAdd(denom + (size_t)dst * B_DIM + b, ex);
}

// wave per (edge, batch): F[b,dst,:] += alpha * XL[b,src,:]
__global__ __launch_bounds__(256)
void edge_scatter_kernel(const float* __restrict__ XL,
                         const long long* __restrict__ ei,
                         const float* __restrict__ ebuf,
                         const float* __restrict__ denom,
                         float* __restrict__ Fbuf,
                         int E, int EN, int N)
{
    int widx = blockIdx.x * (blockDim.x >> 5) + (threadIdx.x >> 5);
    int lane = threadIdx.x & 31;
    if (widx >= EN * B_DIM) return;
    int e = widx >> 2, b = widx & 3;
    int src, dst;
    if (e < E) { src = (int)ei[e]; dst = (int)ei[E + e]; }
    else       { src = dst = e - E; }

    float alpha = ebuf[widx] / denom[(size_t)dst * B_DIM + b];
    const float* hs = XL  + ((size_t)b * N + src) * F_DIM;
    float*      out = Fbuf + ((size_t)b * N + dst) * F_DIM;
#pragma unroll
    for (int j = 0; j < 4; ++j) {
        int f = lane + 32 * j;
        atomicAdd(out + f, hs[f] * alpha);
    }
}

// CAT[row, 0:128]   = bf16(relu(F + bias))
// CAT[row, 128:256] = bf16(H)  or  bf16(H * R)   (useR for gate h)
__global__ void concat_kernel(const float* __restrict__ Fbuf,
                              const float* __restrict__ bias,
                              const float* __restrict__ H,
                              const float* __restrict__ R,
                              __bf16* __restrict__ CAT,
                              int M, int useR)
{
    int i = blockIdx.x * blockDim.x + threadIdx.x;
    if (i >= M * F_DIM) return;
    int f   = i & (F_DIM - 1);
    int row = i >> 7;
    float v = Fbuf[i] + bias[f];
    v = v > 0.f ? v : 0.f;
    float h = H[i];
    if (useR) h *= R[i];
    CAT[(size_t)row * 256 + f]         = (__bf16)v;
    CAT[(size_t)row * 256 + F_DIM + f] = (__bf16)h;
}

// out = Z*H + (1-Z)*Htilde  (Z currently in out)
__global__ void final_kernel(float* __restrict__ out,
                             const float* __restrict__ H,
                             const float* __restrict__ Ht, int n)
{
    int i = blockIdx.x * blockDim.x + threadIdx.x;
    if (i >= n) return;
    float z = out[i];
    out[i] = z * H[i] + (1.f - z) * Ht[i];
}

// ---------------------------------------------------------------------------
extern "C" void kernel_launch(void* const* d_in, const int* in_sizes, int n_in,
                              void* d_out, int out_size, void* d_ws, size_t ws_size,
                              hipStream_t stream)
{
    const float*     X  = (const float*)d_in[0];
    const long long* EI = (const long long*)d_in[1];
    const float*     H  = (const float*)d_in[2];

    const int E  = in_sizes[1] / 2;
    const int M  = in_sizes[0] / F_DIM;   // B*N rows
    const int N  = M / B_DIM;
    const int EN = E + N;

    // workspace carve (256B aligned)
    char* p = (char*)d_ws;
    auto carve = [&](size_t bytes) -> char* {
        char* r = p;
        p += (bytes + 255) & ~(size_t)255;
        return r;
    };
    __bf16* Xbf = (__bf16*)carve((size_t)M * F_DIM * 2);
    __bf16* Wbf[3][3];
    for (int g = 0; g < 3; ++g) {
        Wbf[g][0] = (__bf16*)carve(128 * 128 * 2);   // Wl
        Wbf[g][1] = (__bf16*)carve(128 * 128 * 2);   // Wr
        Wbf[g][2] = (__bf16*)carve(128 * 256 * 2);   // Wlin
    }
    float*  XLb  = (float*)carve((size_t)M * F_DIM * 4);
    float*  XRb  = (float*)carve((size_t)M * F_DIM * 4);
    float*  Fbuf = (float*)carve((size_t)M * F_DIM * 4);
    float*  Rbuf = (float*)carve((size_t)M * F_DIM * 4);
    float*  Htb  = (float*)carve((size_t)M * F_DIM * 4);
    __bf16* CAT  = (__bf16*)carve((size_t)M * 256 * 2);
    float*  ebuf = (float*)carve((size_t)EN * B_DIM * 4);
    float*  mbuf = (float*)carve((size_t)N * B_DIM * 4);
    float*  dbuf = (float*)carve((size_t)N * B_DIM * 4);

    const int TPB = 256;
    auto blocks = [&](int n) { return (n + TPB - 1) / TPB; };

    // --- convert X + weights to bf16 ---
    cvt_bf16_kernel<<<blocks(M * F_DIM), TPB, 0, stream>>>(Xbf, X, M * F_DIM);
    for (int g = 0; g < 3; ++g) {
        int base = 3 + g * 8;
        cvt_bf16_kernel<<<blocks(128 * 128), TPB, 0, stream>>>(
            Wbf[g][0], (const float*)d_in[base + 0], 128 * 128);
        cvt_bf16_kernel<<<blocks(128 * 128), TPB, 0, stream>>>(
            Wbf[g][1], (const float*)d_in[base + 2], 128 * 128);
        cvt_bf16_kernel<<<blocks(128 * 256), TPB, 0, stream>>>(
            Wbf[g][2], (const float*)d_in[base + 6], 128 * 256);
    }

    const int gemmGrid  = (M + 63) / 64;         // 64 rows per block
    const int waveTasks = EN * B_DIM;            // one wave each
    const int waveGrid  = (waveTasks + 7) / 8;   // 8 waves / block

    for (int g = 0; g < 3; ++g) {
        int base = 3 + g * 8;
        const float* bl   = (const float*)d_in[base + 1];
        const float* br   = (const float*)d_in[base + 3];
        const float* att  = (const float*)d_in[base + 4];
        const float* bias = (const float*)d_in[base + 5];
        const float* blin = (const float*)d_in[base + 7];

        // projections (WMMA)
        wmma_gemm_bias_act<<<gemmGrid, TPB, 0, stream>>>(Xbf, Wbf[g][0], bl, XLb, M, 128, 0);
        wmma_gemm_bias_act<<<gemmGrid, TPB, 0, stream>>>(Xbf, Wbf[g][1], br, XRb, M, 128, 0);

        // reset segment buffers
        fill_kernel<<<blocks(N * B_DIM), TPB, 0, stream>>>(mbuf, -3.402823466e38f, N * B_DIM);
        fill_kernel<<<blocks(N * B_DIM), TPB, 0, stream>>>(dbuf, 0.f, N * B_DIM);
        fill_kernel<<<blocks(M * F_DIM), TPB, 0, stream>>>(Fbuf, 0.f, M * F_DIM);

        // attention softmax + aggregate
        edge_score_kernel  <<<waveGrid, TPB, 0, stream>>>(XLb, XRb, EI, att, ebuf, mbuf, E, EN, N);
        edge_expsum_kernel <<<blocks(waveTasks), TPB, 0, stream>>>(EI, ebuf, mbuf, dbuf, E, EN, N);
        edge_scatter_kernel<<<waveGrid, TPB, 0, stream>>>(XLb, EI, ebuf, dbuf, Fbuf, E, EN, N);

        // [relu(F+bias), Hcat] -> bf16
        concat_kernel<<<blocks(M * F_DIM), TPB, 0, stream>>>(
            Fbuf, bias, H, Rbuf, CAT, M, (g == 2) ? 1 : 0);

        // gate GEMM (WMMA, K = 256) with fused activation
        float* dst = (g == 0) ? (float*)d_out : (g == 1) ? Rbuf : Htb;
        int act    = (g == 2) ? 2 : 1;
        wmma_gemm_bias_act<<<gemmGrid, TPB, 0, stream>>>(CAT, Wbf[g][2], blin, dst, M, 256, act);
    }

    // out = Z*H + (1-Z)*Htilde   (Z already in d_out)
    final_kernel<<<blocks(M * F_DIM), TPB, 0, stream>>>((float*)d_out, H, Htb, M * F_DIM);
}
